// Decoder_20796231647398
// MI455X (gfx1250) — matrix operand order repaired and checked by reading, not compiled, not run on hardware
//
#include <hip/hip_runtime.h>
#include <hip/hip_bf16.h>
#include <math.h>

// ---------------------------------------------------------------------------
// Pointer-network decoder for MI455X (gfx1250, wave32, WMMA).
// Persistent cooperative kernel: 16 WGs x 256 threads, 2 grid barriers/step.
// Gate GEMM: fused [32,512]x[512,1024] in f16 via v_wmma_f32_16x16x32_f16,
// f32 accumulation. Weights pre-tiled once into WMMA B-fragment layout.
// ---------------------------------------------------------------------------

typedef __attribute__((ext_vector_type(16))) _Float16 v16h;
typedef __attribute__((ext_vector_type(8)))  _Float16 v8h;
typedef __attribute__((ext_vector_type(8)))  float    v8f;

#define BB   32
#define SS   1024
#define HH   256
#define KTOT 512          // concat(x,h) K dim
#define NWG  16

// workspace byte offsets
#define WS_CNT   0        // unsigned barrier counter
#define WS_BIAS  1024     // 1024 f32  (b_ih + b_hh)
#define WS_XH16  8192     // 32 x 512 f16  (concat x,h — GEMM A operand)
#define WS_GATES 40960    // 32 x 1024 f32
#define WS_WT    172032   // 64nt x 16kt x 32lane x 16halfs f16 (1 MB) B tiles

// ---------------------------------------------------------------------------
// Prep: convert+tile weights to WMMA B-fragment layout, fuse biases,
// zero xh16 (x0=h0=0), reset barrier counter. Runs every launch (deterministic).
// B-layout (32x16 KxN f16 frag): lane L holds column N=L&15; halfs 0..7 hold
// K-base+0..7, halfs 8..15 K-base+16..23, K-base = kt*32 + (L>=16 ? 8 : 0).
// ---------------------------------------------------------------------------
__global__ void prep_kernel(const float* __restrict__ W_ih,
                            const float* __restrict__ W_hh,
                            const float* __restrict__ b_ih,
                            const float* __restrict__ b_hh,
                            _Float16* __restrict__ wtiled,
                            float* __restrict__ bias,
                            _Float16* __restrict__ xh16,
                            unsigned* __restrict__ cnt) {
  int tid  = blockIdx.x * 256 + threadIdx.x;   // [0, 32768)
  int lane = tid & 31;
  int kt   = (tid >> 5) & 15;
  int nt   = tid >> 9;                          // [0, 64)
  int j    = nt * 16 + (lane & 15);             // gate column (N)
  int k0   = kt * 32 + ((lane >> 4) ? 8 : 0);   // K base for this lane
  _Float16* dst = wtiled + ((size_t)((nt * 16 + kt) * 32 + lane)) * 16;
  #pragma unroll
  for (int q = 0; q < 16; ++q) {
    int k = k0 + (q & 7) + ((q >> 3) << 4);
    float w = (k < HH) ? W_ih[j * HH + k] : W_hh[j * HH + (k - HH)];
    dst[q] = (_Float16)w;
  }
  if (tid < 1024)        bias[tid] = b_ih[tid] + b_hh[tid];
  if (tid < BB * KTOT)   xh16[tid] = (_Float16)0.0f;
  if (tid == 0)          *cnt = 0u;
}

// ---------------------------------------------------------------------------
// Grid-wide barrier: monotonic counter, device-scope atomics, sense-free.
// ---------------------------------------------------------------------------
__device__ __forceinline__ void grid_barrier(unsigned* cnt, unsigned target) {
  __threadfence();            // release all prior global writes to device scope
  __syncthreads();
  if (threadIdx.x == 0) {
    __hip_atomic_fetch_add(cnt, 1u, __ATOMIC_ACQ_REL, __HIP_MEMORY_SCOPE_AGENT);
    while (__hip_atomic_load(cnt, __ATOMIC_ACQUIRE, __HIP_MEMORY_SCOPE_AGENT) < target)
      __builtin_amdgcn_s_sleep(2);
  }
  __syncthreads();
  __threadfence();            // acquire: invalidate stale near caches
}

__device__ __forceinline__ float sigmoidf_(float x) {
  return 1.0f / (1.0f + expf(-x));
}

// ---------------------------------------------------------------------------
// Persistent decoder kernel. 16 WGs x 256 threads (8 waves).
// ---------------------------------------------------------------------------
__launch_bounds__(256, 1)
__global__ void decoder_kernel(const float* __restrict__ enc,      // [32,1024,256]
                               const _Float16* __restrict__ wtiled,
                               const float* __restrict__ bias,     // [1024]
                               _Float16* __restrict__ xh16,        // [32,512]
                               float* __restrict__ gates,          // [32,1024]
                               unsigned* __restrict__ cnt,
                               float* __restrict__ out) {          // [32,1024,1024]
  __shared__ float hlds[2][HH];      // h for this WG's 2 batch rows
  __shared__ float sc[2][SS];        // attention scores / exp scratch
  __shared__ float redv[256];
  __shared__ int   redi[256];
  __shared__ float reds[256];

  const int t    = threadIdx.x;
  const int wave = t >> 5;
  const int lane = t & 31;

  // --- GEMM tile assignment: one 16x16 output tile per wave (128 tiles) ---
  const int tile = blockIdx.x * 8 + wave;       // [0,128)
  const int mt   = tile >> 6;                   // [0,2)  batch tile
  const int nt   = tile & 63;                   // [0,64) gate-col tile
  // A-operand lane mapping (16-bit A 16x32 layout per ISA 7.12.2)
  const int arow = mt * 16 + (lane & 15);       // batch row
  const int ahi  = lane >> 4;                   // 0: K {0..7,16..23}; 1: +8
  // D-operand lane mapping (f32 C/D 16x16 layout)
  const int jcol  = nt * 16 + (lane & 15);
  const int mrow0 = mt * 16 + (ahi ? 8 : 0);
  const float bj  = bias[jcol];

  const v16h* bwt = ((const v16h*)wtiled) + (size_t)(nt * 16) * 32 + lane;

  float creg[2] = {0.0f, 0.0f};                 // cell state, 2 batch rows/WG
  unsigned epoch = 0;

  for (int step = 0; step < SS; ++step) {
    // ================= Phase 1: gate GEMM (all waves, WMMA) ================
    v8f acc = {};
    const _Float16* aprow = xh16 + arow * KTOT + ahi * 8;
    #pragma unroll 4
    for (int kt = 0; kt < 16; ++kt) {
      __builtin_prefetch((const void*)(bwt + (kt + 1) * 32), 0, 0);
      v8h alo = *(const v8h*)(aprow + kt * 32);        // K = base+0..7
      v8h ahh = *(const v8h*)(aprow + kt * 32 + 16);   // K = base+16..23
      v16h a = __builtin_shufflevector(alo, ahh, 0, 1, 2, 3, 4, 5, 6, 7,
                                       8, 9, 10, 11, 12, 13, 14, 15);
      v16h b = bwt[(size_t)kt * 32];
      acc = __builtin_amdgcn_wmma_f32_16x16x32_f16(
          false, a, false, b, (short)0, acc, false, false);
    }
    #pragma unroll
    for (int r = 0; r < 8; ++r)
      gates[(mrow0 + r) * 1024 + jcol] = acc[r] + bj;

    grid_barrier(cnt, NWG * (++epoch));   // gates visible everywhere

    // ========== Phase 2: LSTM cell for this WG's 2 batch rows ==============
    #pragma unroll
    for (int bl = 0; bl < 2; ++bl) {
      const int b = blockIdx.x * 2 + bl;
      const float* g = gates + b * 1024;
      float ig = sigmoidf_(g[t]);           // i,f,g,o PyTorch gate order
      float fg = sigmoidf_(g[t + 256]);
      float gg = tanhf(g[t + 512]);
      float og = sigmoidf_(g[t + 768]);
      float cc = fg * creg[bl] + ig * gg;
      creg[bl] = cc;
      float hh = og * tanhf(cc);
      hlds[bl][t] = hh;
      xh16[b * KTOT + HH + t] = (_Float16)hh;   // h half of next A operand
    }
    __syncthreads();

    // ========== Phase 3: attention scores (coalesced, lane-split K) ========
    {
      const int bl = wave >> 2;                  // waves 0-3 -> b0, 4-7 -> b1
      const int b  = blockIdx.x * 2 + bl;
      const int s0 = (wave & 3) << 8;            // 256 s per wave
      for (int si = 0; si < 256; ++si) {
        const int s = s0 + si;
        const float* er = enc + ((size_t)b * SS + s) * HH + lane;
        float p = 0.0f;
        #pragma unroll
        for (int j = 0; j < 8; ++j)
          p += er[32 * j] * hlds[bl][lane + 32 * j];
        #pragma unroll
        for (int off = 16; off; off >>= 1)
          p += __shfl_xor(p, off, 32);
        if (lane == 0) sc[bl][s] = p;
      }
    }
    __syncthreads();

    // ========== Phase 4: softmax + argmax + write probs + gather x =========
    for (int bl = 0; bl < 2; ++bl) {
      const int b = blockIdx.x * 2 + bl;
      // max + first-occurrence argmax
      float mv = -INFINITY; int mi = 0x7fffffff;
      for (int s = t; s < SS; s += 256) {
        float v = sc[bl][s];
        if (v > mv) { mv = v; mi = s; }
      }
      redv[t] = mv; redi[t] = mi;
      __syncthreads();
      for (int off = 128; off; off >>= 1) {
        if (t < off) {
          float v2 = redv[t + off]; int i2 = redi[t + off];
          if (v2 > redv[t] || (v2 == redv[t] && i2 < redi[t])) {
            redv[t] = v2; redi[t] = i2;
          }
        }
        __syncthreads();
      }
      const float mx  = redv[0];
      const int  midx = redi[0];
      // exp + sum
      float lsum = 0.0f;
      for (int s = t; s < SS; s += 256) {
        float e = expf(sc[bl][s] - mx);
        sc[bl][s] = e;
        lsum += e;
      }
      reds[t] = lsum;
      __syncthreads();
      for (int off = 128; off; off >>= 1) {
        if (t < off) reds[t] += reds[t + off];
        __syncthreads();
      }
      const float inv = 1.0f / reds[0];
      float* orow = out + ((size_t)b * SS + step) * SS;
      for (int s = t; s < SS; s += 256) orow[s] = sc[bl][s] * inv;
      // pointer gather: x_next = enc[b, argmax, :]  -> x half of A operand
      xh16[b * KTOT + t] = (_Float16)enc[((size_t)b * SS + midx) * HH + t];
      __syncthreads();   // protect redv/redi/reds reuse next bl
    }

    grid_barrier(cnt, NWG * (++epoch));   // xh16 (x and h halves) visible
  }
}

// ---------------------------------------------------------------------------
extern "C" void kernel_launch(void* const* d_in, const int* in_sizes, int n_in,
                              void* d_out, int out_size, void* d_ws, size_t ws_size,
                              hipStream_t stream) {
  (void)in_sizes; (void)n_in; (void)out_size; (void)ws_size;
  const float* enc  = (const float*)d_in[0];   // [32,1024,256]
  const float* W_ih = (const float*)d_in[1];   // [1024,256]
  const float* W_hh = (const float*)d_in[2];   // [1024,256]
  const float* b_ih = (const float*)d_in[3];   // [1024]
  const float* b_hh = (const float*)d_in[4];   // [1024]
  float* out = (float*)d_out;                  // [32,1024,1024]

  char* ws = (char*)d_ws;
  unsigned* cnt    = (unsigned*)(ws + WS_CNT);
  float*    bias   = (float*)(ws + WS_BIAS);
  _Float16* xh16   = (_Float16*)(ws + WS_XH16);
  float*    gates  = (float*)(ws + WS_GATES);
  _Float16* wtiled = (_Float16*)(ws + WS_WT);

  prep_kernel<<<128, 256, 0, stream>>>(W_ih, W_hh, b_ih, b_hh,
                                       wtiled, bias, xh16, cnt);
  decoder_kernel<<<NWG, 256, 0, stream>>>(enc, wtiled, bias, xh16,
                                          gates, cnt, out);
}
